// SelfAttnBlock_46797963657495
// MI455X (gfx1250) — matrix operand order repaired and checked by reading, not compiled, not run on hardware
//
#include <hip/hip_runtime.h>
#include <hip/hip_bf16.h>
#include <math.h>

typedef __attribute__((ext_vector_type(16))) __bf16 v16bf;
typedef __attribute__((ext_vector_type(8)))  __bf16 v8bf;
typedef __attribute__((ext_vector_type(8)))  float  v8f;

namespace {

constexpr int Bn   = 2;
constexpr int Cch  = 256;
constexpr int Him  = 128;
constexpr int Wim  = 128;
constexpr int Npx  = Him * Wim;     // 16384
constexpr int Mrow = Bn * Npx;      // 32768
constexpr float EPS = 1e-5f;

// ---------------- weight pack: W (Nout x K) f32 -> Wt (K x Nout) bf16 ----------------
__global__ void pack_wT_kernel(const float* __restrict__ W, __bf16* __restrict__ Wt,
                               int Nout, int K) {
  int i = blockIdx.x * blockDim.x + threadIdx.x;
  if (i < Nout * K) {
    int k = i / Nout;
    int n = i - k * Nout;
    Wt[i] = (__bf16)W[(size_t)n * K + k];
  }
}

// ---------------- LN over C of x (B,C,H,W) -> q bf16 (B*N, C), via LDS tile ----------
__global__ __launch_bounds__(256) void ln_chw_kernel(
    const float* __restrict__ x, const float* __restrict__ g,
    const float* __restrict__ bta, __bf16* __restrict__ q) {
  __shared__ float tile[32][Cch + 1];
  const int tid = threadIdx.x;
  const int b  = blockIdx.x >> 9;               // N/32 = 512 blocks per batch
  const int n0 = (blockIdx.x & 511) * 32;
  {
    const int nsub = tid & 31;
    const int cg   = tid >> 5;                  // 0..7
    for (int i = 0; i < 32; ++i) {
      int c = cg * 32 + i;
      tile[nsub][c] = x[((size_t)b * Cch + c) * Npx + n0 + nsub];
    }
  }
  __syncthreads();
  const int p  = tid >> 3;                      // pixel within tile (0..31)
  const int gi = tid & 7;                       // 8 lanes per pixel
  float s = 0.f, s2 = 0.f;
  for (int i = 0; i < 32; ++i) {
    float v = tile[p][gi * 32 + i];
    s += v; s2 += v * v;
  }
  s  += __shfl_xor(s, 1, 32);  s  += __shfl_xor(s, 2, 32);  s  += __shfl_xor(s, 4, 32);
  s2 += __shfl_xor(s2, 1, 32); s2 += __shfl_xor(s2, 2, 32); s2 += __shfl_xor(s2, 4, 32);
  const float mean = s * (1.f / Cch);
  const float var  = s2 * (1.f / Cch) - mean * mean;
  const float rstd = rsqrtf(var + EPS);
  const size_t row = (size_t)b * Npx + n0 + p;
  for (int i = 0; i < 32; ++i) {
    int c = gi * 32 + i;
    float v = (tile[p][c] - mean) * rstd * g[c] + bta[c];
    q[row * Cch + c] = (__bf16)v;
  }
}

// ---------------- LN over C of row-major (B*N, C) -> transposed f32 (B,C,N) ----------
__global__ __launch_bounds__(256) void ln_rowmajor_T_kernel(
    const float* __restrict__ in, const float* __restrict__ g,
    const float* __restrict__ bta, float* __restrict__ yT) {
  const int lane = threadIdx.x & 31;
  const int pm   = blockIdx.x * 8 + (threadIdx.x >> 5);   // pixel (b*N+n)
  const int b = pm >> 14;
  const int n = pm & (Npx - 1);
  float v[8];
  float s = 0.f, s2 = 0.f;
  for (int j = 0; j < 8; ++j) {
    int c = lane + 32 * j;
    float t = in[(size_t)pm * Cch + c];
    v[j] = t; s += t; s2 += t * t;
  }
  for (int m = 1; m < 32; m <<= 1) { s += __shfl_xor(s, m, 32); s2 += __shfl_xor(s2, m, 32); }
  const float mean = s * (1.f / Cch);
  const float rstd = rsqrtf(s2 * (1.f / Cch) - mean * mean + EPS);
  for (int j = 0; j < 8; ++j) {
    int c = lane + 32 * j;
    yT[((size_t)b * Cch + c) * Npx + n] = (v[j] - mean) * rstd * g[c] + bta[c];
  }
}

// ---------------- WMMA bf16 GEMM: C = A(MxK) * Wt(KxN) + bias (+residual modes) ------
__device__ __forceinline__ v16bf load_a_frag(const __bf16* __restrict__ A, int lda,
                                             int row, int k0, int lane) {
  const __bf16* p0 = A + (size_t)row * lda + k0 + ((lane & 16) ? 8 : 0);
  v8bf lo = *(const v8bf*)p0;
  v8bf hi = *(const v8bf*)(p0 + 16);
  return __builtin_shufflevector(lo, hi, 0,1,2,3,4,5,6,7,8,9,10,11,12,13,14,15);
}

__device__ __forceinline__ v16bf load_b_frag(const __bf16* __restrict__ Wt, int Nout,
                                             int k0, int n0, int lane) {
  const __bf16* p0 = Wt + (size_t)(k0 + ((lane & 16) ? 8 : 0)) * Nout + n0 + (lane & 15);
  v16bf r;
  for (int s = 0; s < 8; ++s) {
    r[s]     = p0[(size_t)s * Nout];
    r[s + 8] = p0[(size_t)(s + 16) * Nout];
  }
  return r;
}

__device__ __forceinline__ void gemm_store(v8f acc, int m0, int ncolBase, int lane,
                                           int Nout, int mode,
                                           const float* __restrict__ bias,
                                           float* __restrict__ Cout,
                                           const float* __restrict__ res,
                                           int NpixA, int Cres) {
  const int ncol = ncolBase + (lane & 15);
  if (ncol >= Nout) return;
  const float bv = bias[ncol];
  const int mBase = m0 + ((lane & 16) ? 8 : 0);
  for (int r = 0; r < 8; ++r) {
    int m = mBase + r;
    float v = acc[r] + bv;
    if (mode == 1) {              // + x transposed residual, row-major store
      int bb = m / NpixA, hw = m - bb * NpixA;
      v += res[((size_t)bb * Cres + ncol) * NpixA + hw];
      Cout[(size_t)m * Nout + ncol] = v;
    } else if (mode == 2) {       // + row-major residual, transposed (B,C,N) store
      int bb = m / NpixA, hw = m - bb * NpixA;
      v += res[(size_t)m * Cres + ncol];
      Cout[((size_t)bb * Nout + ncol) * NpixA + hw] = v;
    } else {
      Cout[(size_t)m * Nout + ncol] = v;
    }
  }
}

__global__ __launch_bounds__(256) void gemm_bf16_wmma(
    const __bf16* __restrict__ A, const __bf16* __restrict__ Wt,
    const float* __restrict__ bias, float* __restrict__ Cout,
    int M, int Nout, int K, int mode,
    const float* __restrict__ res, int NpixA, int Cres) {
  const int tid  = threadIdx.x;
  const int lane = tid & 31;
  const int wid  = tid >> 5;                       // 8 waves: 4 in M, 2 in N
  const int m0   = blockIdx.y * 64 + (wid >> 1) * 16;
  const int n0w  = blockIdx.x * 128 + (wid & 1) * 64;
  const int arow = m0 + (lane & 15);

  v8f acc0 = {}, acc1 = {}, acc2 = {}, acc3 = {};
  const int nb0 = (n0w + 16 <= Nout) ? n0w      : 0;
  const int nb1 = (n0w + 32 <= Nout) ? n0w + 16 : 0;
  const int nb2 = (n0w + 48 <= Nout) ? n0w + 32 : 0;
  const int nb3 = (n0w + 64 <= Nout) ? n0w + 48 : 0;

  for (int k0 = 0; k0 < K; k0 += 32) {
    v16bf a  = load_a_frag(A, K, arow, k0, lane);
    v16bf b0 = load_b_frag(Wt, Nout, k0, nb0, lane);
    v16bf b1 = load_b_frag(Wt, Nout, k0, nb1, lane);
    v16bf b2 = load_b_frag(Wt, Nout, k0, nb2, lane);
    v16bf b3 = load_b_frag(Wt, Nout, k0, nb3, lane);
    acc0 = __builtin_amdgcn_wmma_f32_16x16x32_bf16(false, a, false, b0, (short)0, acc0, false, false);
    acc1 = __builtin_amdgcn_wmma_f32_16x16x32_bf16(false, a, false, b1, (short)0, acc1, false, false);
    acc2 = __builtin_amdgcn_wmma_f32_16x16x32_bf16(false, a, false, b2, (short)0, acc2, false, false);
    acc3 = __builtin_amdgcn_wmma_f32_16x16x32_bf16(false, a, false, b3, (short)0, acc3, false, false);
  }
  gemm_store(acc0, m0, n0w +  0, lane, Nout, mode, bias, Cout, res, NpixA, Cres);
  gemm_store(acc1, m0, n0w + 16, lane, Nout, mode, bias, Cout, res, NpixA, Cres);
  gemm_store(acc2, m0, n0w + 32, lane, Nout, mode, bias, Cout, res, NpixA, Cres);
  gemm_store(acc3, m0, n0w + 48, lane, Nout, mode, bias, Cout, res, NpixA, Cres);
}

// ---------------- softmax over NP + bilinear deform-sample -> attn_out bf16 ----------
__global__ __launch_bounds__(256) void deform_sample_kernel(
    const float* __restrict__ val, const float* __restrict__ offv,
    const float* __restrict__ logits, __bf16* __restrict__ ao) {
  const int t  = blockIdx.x * blockDim.x + threadIdx.x;
  const int dc = t & 3;                 // 16-channel chunk of D=64
  const int hh = (t >> 2) & 3;          // head
  const int rest = t >> 4;
  const int n = rest & (Npx - 1);
  const int b = rest >> 14;
  const int wpx = n & (Wim - 1);
  const int hpx = n >> 7;
  const size_t pix = (size_t)b * Npx + n;

  float la[4];
  for (int p = 0; p < 4; ++p) la[p] = logits[pix * 16 + hh * 4 + p];
  float mx = fmaxf(fmaxf(la[0], la[1]), fmaxf(la[2], la[3]));
  float aw[4]; float ssum = 0.f;
  for (int p = 0; p < 4; ++p) { aw[p] = expf(la[p] - mx); ssum += aw[p]; }
  const float inv = 1.f / ssum;

  float acc[16];
  for (int i = 0; i < 16; ++i) acc[i] = 0.f;

  for (int p = 0; p < 4; ++p) {
    float ox = offv[pix * 32 + hh * 8 + p * 2 + 0];
    float oy = offv[pix * 32 + hh * 8 + p * 2 + 1];
    float px = (float)wpx + ox;      // (ref + off/norm)*W - 0.5 simplifies to w + off
    float py = (float)hpx + oy;
    float x0f = floorf(px), y0f = floorf(py);
    float wx = px - x0f, wy = py - y0f;
    int x0 = (int)x0f, y0 = (int)y0f;
    float wp = aw[p] * inv;
    float ws4[4] = { (1.f - wx) * (1.f - wy) * wp, wx * (1.f - wy) * wp,
                     (1.f - wx) * wy * wp,         wx * wy * wp };
    int xs[4] = { x0, x0 + 1, x0, x0 + 1 };
    int ys[4] = { y0, y0, y0 + 1, y0 + 1 };
    for (int cnr = 0; cnr < 4; ++cnr) {
      int xx = xs[cnr], yy = ys[cnr];
      if (xx < 0 || xx >= Wim || yy < 0 || yy >= Him) continue;
      const float* vp = val + ((size_t)b * Npx + yy * Wim + xx) * Cch + hh * 64 + dc * 16;
      float w = ws4[cnr];
      for (int i = 0; i < 16; ++i) acc[i] += w * vp[i];
    }
  }
  __bf16* op = ao + pix * Cch + hh * 64 + dc * 16;
  for (int i = 0; i < 16; ++i) op[i] = (__bf16)acc[i];
}

// ---------------- depthwise 3x3 (SAME) + bias + SiLU gate -> h bf16 (B*N, 128) -------
__global__ __launch_bounds__(256) void dwconv_gate_kernel(
    const float* __restrict__ y, const float* __restrict__ kw,
    const float* __restrict__ kb, __bf16* __restrict__ hbf) {
  const int t  = blockIdx.x * blockDim.x + threadIdx.x;
  const int n  = t & (Npx - 1);
  const int c2 = (t >> 14) & 127;
  const int b  = t >> 21;
  const int hpx = n >> 7, wpx = n & (Wim - 1);
  float s[2];
  for (int j = 0; j < 2; ++j) {
    int ch = c2 + j * 128;
    const float* yp = y + ((size_t)b * Cch + ch) * Npx;
    const float* kp = kw + ch * 9;
    float a = kb[ch];
    for (int kh = 0; kh < 3; ++kh) {
      int yy = hpx + kh - 1;
      if (yy < 0 || yy >= Him) continue;
      for (int kx = 0; kx < 3; ++kx) {
        int xx = wpx + kx - 1;
        if (xx < 0 || xx >= Wim) continue;
        a += yp[yy * Wim + xx] * kp[kh * 3 + kx];
      }
    }
    s[j] = a;
  }
  float sig = 1.f / (1.f + expf(-s[0]));
  hbf[((size_t)b * Npx + n) * 128 + c2] = (__bf16)(s[0] * sig * s[1]);
}

} // anonymous namespace

extern "C" void kernel_launch(void* const* d_in, const int* in_sizes, int n_in,
                              void* d_out, int out_size, void* d_ws, size_t ws_size,
                              hipStream_t stream) {
  const float* x      = (const float*)d_in[0];
  const float* norm_g = (const float*)d_in[1];
  const float* norm_b = (const float*)d_in[2];
  const float* ffn_g  = (const float*)d_in[3];
  const float* ffn_b  = (const float*)d_in[4];
  const float* W_val  = (const float*)d_in[5];
  const float* b_val  = (const float*)d_in[6];
  const float* W_off  = (const float*)d_in[7];
  const float* b_off  = (const float*)d_in[8];
  const float* W_attn = (const float*)d_in[9];
  const float* b_attn = (const float*)d_in[10];
  const float* W_out  = (const float*)d_in[11];
  const float* b_out  = (const float*)d_in[12];
  const float* dw_w   = (const float*)d_in[13];
  const float* dw_b   = (const float*)d_in[14];
  const float* pr_w   = (const float*)d_in[15];
  const float* pr_b   = (const float*)d_in[16];
  float* out = (float*)d_out;

  // workspace carve-up (256B aligned)
  char* ws = (char*)d_ws;
  size_t cur = 0;
  auto carve = [&](size_t bytes) -> void* {
    void* p = ws + cur;
    cur = (cur + bytes + 255) & ~(size_t)255;
    return p;
  };
  __bf16* q_bf    = (__bf16*)carve((size_t)Mrow * Cch * 2);
  __bf16* ao_bf   = (__bf16*)carve((size_t)Mrow * Cch * 2);
  __bf16* h_bf    = (__bf16*)carve((size_t)Mrow * 128 * 2);
  __bf16* wvalT   = (__bf16*)carve(256 * 256 * 2);
  __bf16* woffT   = (__bf16*)carve(32 * 256 * 2);
  __bf16* wattnT  = (__bf16*)carve(16 * 256 * 2);
  __bf16* woutT   = (__bf16*)carve(256 * 256 * 2);
  __bf16* prwT    = (__bf16*)carve(256 * 128 * 2);
  float*  val_f   = (float*)carve((size_t)Mrow * Cch * 4);
  float*  off_f   = (float*)carve((size_t)Mrow * 32 * 4);
  float*  attnl_f = (float*)carve((size_t)Mrow * 16 * 4);
  float*  out_f   = (float*)carve((size_t)Mrow * Cch * 4);
  float*  y_f     = (float*)carve((size_t)Mrow * Cch * 4);

  // 1) pack + transpose weights to bf16 (K x Nout)
  pack_wT_kernel<<<(256 * 256 + 255) / 256, 256, 0, stream>>>(W_val, wvalT, 256, 256);
  pack_wT_kernel<<<(32 * 256 + 255) / 256, 256, 0, stream>>>(W_off, woffT, 32, 256);
  pack_wT_kernel<<<(16 * 256 + 255) / 256, 256, 0, stream>>>(W_attn, wattnT, 16, 256);
  pack_wT_kernel<<<(256 * 256 + 255) / 256, 256, 0, stream>>>(W_out, woutT, 256, 256);
  pack_wT_kernel<<<(256 * 128 + 255) / 256, 256, 0, stream>>>(pr_w, prwT, 256, 128);

  // 2) LayerNorm #1 (x -> q bf16)
  ln_chw_kernel<<<Bn * (Npx / 32), 256, 0, stream>>>(x, norm_g, norm_b, q_bf);

  // 3) q @ {W_val, W_off, W_attn}^T + bias (WMMA)
  gemm_bf16_wmma<<<dim3(2, Mrow / 64), 256, 0, stream>>>(q_bf, wvalT, b_val, val_f,
      Mrow, 256, 256, 0, nullptr, Npx, 0);
  gemm_bf16_wmma<<<dim3(1, Mrow / 64), 256, 0, stream>>>(q_bf, woffT, b_off, off_f,
      Mrow, 32, 256, 0, nullptr, Npx, 0);
  gemm_bf16_wmma<<<dim3(1, Mrow / 64), 256, 0, stream>>>(q_bf, wattnT, b_attn, attnl_f,
      Mrow, 16, 256, 0, nullptr, Npx, 0);

  // 4) softmax + bilinear deformable sampling -> attn_out bf16
  deform_sample_kernel<<<(Bn * Npx * 16) / 256, 256, 0, stream>>>(val_f, off_f, attnl_f, ao_bf);

  // 5) out = attn_out @ W_out^T + b_out + xf   (WMMA, mode 1 adds transposed x)
  gemm_bf16_wmma<<<dim3(2, Mrow / 64), 256, 0, stream>>>(ao_bf, woutT, b_out, out_f,
      Mrow, 256, 256, 1, x, Npx, 256);

  // 6) LayerNorm #2 (out -> y, stored transposed (B,C,N) for conv)
  ln_rowmajor_T_kernel<<<Mrow / 8, 256, 0, stream>>>(out_f, ffn_g, ffn_b, y_f);

  // 7) depthwise 3x3 + bias + SiLU gate -> h bf16 (B*N, 128)
  dwconv_gate_kernel<<<(Bn * 128 * Npx) / 256, 256, 0, stream>>>(y_f, dw_w, dw_b, h_bf);

  // 8) final = h @ pr_w^T + pr_b + out  -> stored transposed into d_out (B,C,H,W)
  gemm_bf16_wmma<<<dim3(2, Mrow / 64), 256, 0, stream>>>(h_bf, prwT, pr_b, out,
      Mrow, 256, 128, 2, out_f, Npx, 256);
}
